// Old_TemporalDecoderLSTM_72413148610994
// MI455X (gfx1250) — matrix operand order repaired and checked by reading, not compiled, not run on hardware
//
#include <hip/hip_runtime.h>

// ---------------- problem constants ----------------
#define B_AGENTS 8192
#define IN_LEN   20
#define PRED_LEN 30
#define H_DIM    128
#define E_DIM    64
#define F_IN     40
#define F_PAD    64
#define G_DIM    512      // 4*H
#define RPB      16       // rows per block (M tile)
#define LN_EPS   1e-5f

typedef __bf16 bf16;
typedef __attribute__((ext_vector_type(16))) __bf16 v16bf;
typedef __attribute__((ext_vector_type(8)))  float  v8f;

// ---- WMMA fragment builders (contiguous K order k = 16*(lane/16) + e, same for A and B) ----

// A-matrix 16x32 bf16, one row per lane-group; rowPtr -> LDS row (contiguous K)
static __device__ __forceinline__ v16bf load_a_frag(const bf16* rowPtr, int lane, int kb) {
  const int hi = (lane < 16) ? 0 : 16;
  const bf16* p0 = rowPtr + kb + hi;        // e=0..15 -> K = kb+hi+e
  v16bf a;
#pragma unroll
  for (int e = 0; e < 16; ++e) { a[e] = p0[e]; }
  return a;
}

// B-matrix 32x16: B[k][n] = W[n][k], W row-major [N][ldk]; 16 contiguous bf16/lane
static __device__ __forceinline__ v16bf load_b_frag(const bf16* W, int ldk, int n0, int kb, int lane) {
  const int n  = n0 + (lane & 15);
  const int ks = kb + ((lane < 16) ? 0 : 16);
  const bf16* p = W + (size_t)n * ldk + ks;
  v16bf b;
#pragma unroll
  for (int e = 0; e < 16; ++e) b[e] = p[e];
  return b;
}

static __device__ __forceinline__ float sigmoidf_(float x) {
  return 1.0f / (1.0f + __expf(-x));
}

// ---------------- one-time weight conversion ----------------
__global__ void k_convert(const float* __restrict__ embW, const float* __restrict__ Wih,
                          const float* __restrict__ Whh, const float* __restrict__ bih,
                          const float* __restrict__ bhh,
                          bf16* __restrict__ embp, bf16* __restrict__ wih_b,
                          bf16* __restrict__ whh_b, float* __restrict__ biasc) {
  int i = blockIdx.x * blockDim.x + threadIdx.x;
  const int n_whh = G_DIM * H_DIM;   // 65536
  const int n_wih = G_DIM * E_DIM;   // 32768
  const int n_emb = E_DIM * F_PAD;   // 4096 (K padded 40->64 with zeros)
  if (i < n_whh) { whh_b[i] = (bf16)Whh[i]; return; }
  i -= n_whh;
  if (i < n_wih) { wih_b[i] = (bf16)Wih[i]; return; }
  i -= n_wih;
  if (i < n_emb) {
    int n = i >> 6, k = i & 63;
    embp[i] = (bf16)((k < F_IN) ? embW[n * F_IN + k] : 0.0f);
    return;
  }
  i -= n_emb;
  if (i < G_DIM) biasc[i] = bih[i] + bhh[i];
}

// ---------------- per-step: LN1 + embedding GEMM + leaky ReLU ----------------
// window buffer win[(IN_LEN+PRED_LEN)*B*2]; feature f of row i at step t = win[t*16384 + i*40 + f]
__global__ void k_embed(const float* __restrict__ win, const float* __restrict__ ln1w,
                        const float* __restrict__ ln1b, const bf16* __restrict__ embp,
                        const float* __restrict__ embb, bf16* __restrict__ x, int t) {
  __shared__ bf16 sFeat[RPB][F_PAD];
  const int rowBase = blockIdx.x * RPB;
  const int tid = threadIdx.x;           // 128 threads = 4 waves

  if (tid < RPB) {                       // serial LN over 40 feats per row
    const float* p = win + (size_t)t * (B_AGENTS * 2) + (size_t)(rowBase + tid) * F_IN;
    float v[F_IN], s = 0.f, s2 = 0.f;
#pragma unroll
    for (int f = 0; f < F_IN; ++f) { v[f] = p[f]; s += v[f]; s2 += v[f] * v[f]; }
    const float mean = s * (1.0f / F_IN);
    const float var  = s2 * (1.0f / F_IN) - mean * mean;
    const float rs   = rsqrtf(var + LN_EPS);
#pragma unroll
    for (int f = 0; f < F_IN; ++f)
      sFeat[tid][f] = (bf16)(((v[f] - mean) * rs) * ln1w[f] + ln1b[f]);
#pragma unroll
    for (int f = F_IN; f < F_PAD; ++f) sFeat[tid][f] = (bf16)0.0f;
  }
  __syncthreads();

  const int lane = tid & 31;
  const int wave = tid >> 5;             // 0..3 -> N tile
  const int n0   = wave * 16;
  const int m    = lane & 15;
  v8f acc = {};
#pragma unroll
  for (int kb = 0; kb < F_PAD; kb += 32) {
    v16bf a = load_a_frag(&sFeat[m][0], lane, kb);
    v16bf b = load_b_frag(embp, F_PAD, n0, kb, lane);
    acc = __builtin_amdgcn_wmma_f32_16x16x32_bf16(false, a, false, b, (short)0, acc, false, false);
  }
  const int n     = n0 + (lane & 15);
  const int mBase = (lane < 16) ? 0 : 8;
  const float bias = embb[n];
#pragma unroll
  for (int r = 0; r < 8; ++r) {
    float g = acc[r] + bias;
    g = (g > 0.f) ? g : 0.01f * g;       // leaky_relu(0.01)
    x[(size_t)(rowBase + mBase + r) * E_DIM + n] = (bf16)g;
  }
}

// ---------------- per-step: gates GEMM + LSTM cell + LN2 + pos ----------------
__global__ void k_cell(const bf16* __restrict__ x, const bf16* __restrict__ wih,
                       const bf16* __restrict__ whh, const float* __restrict__ biasc,
                       float* __restrict__ Hbuf, float* __restrict__ Cbuf,
                       const float* __restrict__ ln2w, const float* __restrict__ ln2b,
                       const float* __restrict__ posW, const float* __restrict__ posb,
                       float* __restrict__ out, float* __restrict__ win, int t) {
  __shared__ bf16  sX[RPB][E_DIM];     //  2 KB
  __shared__ bf16  sH[RPB][H_DIM];     //  4 KB
  __shared__ float sG[RPB][G_DIM];     // 32 KB
  __shared__ float sHn[RPB][H_DIM];    //  8 KB
  const int rowBase = blockIdx.x * RPB;
  const int tid = threadIdx.x;         // 256 threads = 8 waves

  for (int i = tid; i < RPB * E_DIM; i += 256) {
    int mm = i >> 6, k = i & 63;
    sX[mm][k] = x[(size_t)(rowBase + mm) * E_DIM + k];
  }
  for (int i = tid; i < RPB * H_DIM; i += 256) {
    int mm = i >> 7, k = i & 127;
    sH[mm][k] = (bf16)Hbuf[(size_t)(rowBase + mm) * H_DIM + k];
  }
  __syncthreads();

  const int lane  = tid & 31;
  const int wave  = tid >> 5;          // 0..7 -> 64-gate-column chunk
  const int m     = lane & 15;
  const int mBase = (lane < 16) ? 0 : 8;
#pragma unroll
  for (int nt = 0; nt < 4; ++nt) {
    const int n0 = wave * 64 + nt * 16;
    v8f acc = {};
#pragma unroll
    for (int kb = 0; kb < E_DIM; kb += 32) {    // x @ W_ih^T
      v16bf a = load_a_frag(&sX[m][0], lane, kb);
      v16bf b = load_b_frag(wih, E_DIM, n0, kb, lane);
      acc = __builtin_amdgcn_wmma_f32_16x16x32_bf16(false, a, false, b, (short)0, acc, false, false);
    }
#pragma unroll
    for (int kb = 0; kb < H_DIM; kb += 32) {    // h @ W_hh^T
      v16bf a = load_a_frag(&sH[m][0], lane, kb);
      v16bf b = load_b_frag(whh, H_DIM, n0, kb, lane);
      acc = __builtin_amdgcn_wmma_f32_16x16x32_bf16(false, a, false, b, (short)0, acc, false, false);
    }
    const int n = n0 + (lane & 15);
#pragma unroll
    for (int r = 0; r < 8; ++r) sG[mBase + r][n] = acc[r];
  }
  __syncthreads();

  // fused LSTM cell (fp32 state), torch gate order i,f,g,o
#pragma unroll
  for (int e = 0; e < 8; ++e) {
    int lin = tid + 256 * e;           // 0..2047
    int mm = lin >> 7, j = lin & 127;
    const size_t idx = (size_t)(rowBase + mm) * H_DIM + j;
    float gi = sG[mm][j]       + biasc[j];
    float gf = sG[mm][128 + j] + biasc[128 + j];
    float gg = sG[mm][256 + j] + biasc[256 + j];
    float go = sG[mm][384 + j] + biasc[384 + j];
    float cN = sigmoidf_(gf) * Cbuf[idx] + sigmoidf_(gi) * tanhf(gg);
    float hN = sigmoidf_(go) * tanhf(cN);
    Cbuf[idx] = cN;
    Hbuf[idx] = hN;
    sHn[mm][j] = hN;
  }
  __syncthreads();

  // LN2 + hidden2pos; write preds[t] and append to the window buffer
  if (tid < RPB) {
    const int row = rowBase + tid;
    float s = 0.f, s2 = 0.f;
#pragma unroll
    for (int j = 0; j < H_DIM; ++j) { float v = sHn[tid][j]; s += v; s2 += v * v; }
    const float mean = s * (1.0f / H_DIM);
    const float var  = s2 * (1.0f / H_DIM) - mean * mean;
    const float rs   = rsqrtf(var + LN_EPS);
    float d0 = 0.f, d1 = 0.f;
#pragma unroll
    for (int j = 0; j < H_DIM; ++j) {
      float nv = (sHn[tid][j] - mean) * rs * ln2w[j] + ln2b[j];
      d0 += nv * posW[j];              // posW row 0
      d1 += nv * posW[H_DIM + j];      // posW row 1
    }
    d0 += posb[0]; d1 += posb[1];
    out[(size_t)t * (B_AGENTS * 2) + row * 2 + 0] = d0;
    out[(size_t)t * (B_AGENTS * 2) + row * 2 + 1] = d1;
    win[(size_t)(IN_LEN + t) * (B_AGENTS * 2) + row * 2 + 0] = d0;
    win[(size_t)(IN_LEN + t) * (B_AGENTS * 2) + row * 2 + 1] = d1;
  }
}

extern "C" void kernel_launch(void* const* d_in, const int* in_sizes, int n_in,
                              void* d_out, int out_size, void* d_ws, size_t ws_size,
                              hipStream_t stream) {
  (void)in_sizes; (void)n_in; (void)out_size; (void)ws_size;
  const float* traj_rel = (const float*)d_in[1];
  const float* h0   = (const float*)d_in[2];
  const float* c0   = (const float*)d_in[3];
  const float* ln1w = (const float*)d_in[4];
  const float* ln1b = (const float*)d_in[5];
  const float* embW = (const float*)d_in[6];
  const float* embb = (const float*)d_in[7];
  const float* Wih  = (const float*)d_in[8];
  const float* Whh  = (const float*)d_in[9];
  const float* bih  = (const float*)d_in[10];
  const float* bhh  = (const float*)d_in[11];
  const float* ln2w = (const float*)d_in[12];
  const float* ln2b = (const float*)d_in[13];
  const float* posW = (const float*)d_in[14];
  const float* posb = (const float*)d_in[15];

  char* ws = (char*)d_ws;
  size_t off = 0;
  auto carve = [&](size_t bytes) -> char* {
    char* p = ws + off;
    off += (bytes + 255) & ~(size_t)255;
    return p;
  };
  bf16*  whh_b = (bf16*)carve((size_t)G_DIM * H_DIM * 2);
  bf16*  wih_b = (bf16*)carve((size_t)G_DIM * E_DIM * 2);
  bf16*  embp  = (bf16*)carve((size_t)E_DIM * F_PAD * 2);
  float* biasc = (float*)carve((size_t)G_DIM * 4);
  float* win   = (float*)carve((size_t)(IN_LEN + PRED_LEN) * B_AGENTS * 2 * 4);
  bf16*  xbuf  = (bf16*)carve((size_t)B_AGENTS * E_DIM * 2);
  float* Hb    = (float*)carve((size_t)B_AGENTS * H_DIM * 4);
  float* Cb    = (float*)carve((size_t)B_AGENTS * H_DIM * 4);

  const int convTotal = G_DIM * H_DIM + G_DIM * E_DIM + E_DIM * F_PAD + G_DIM;
  k_convert<<<(convTotal + 255) / 256, 256, 0, stream>>>(embW, Wih, Whh, bih, bhh,
                                                         embp, wih_b, whh_b, biasc);
  hipMemcpyAsync(win, traj_rel, (size_t)IN_LEN * B_AGENTS * 2 * 4, hipMemcpyDeviceToDevice, stream);
  hipMemcpyAsync(Hb, h0, (size_t)B_AGENTS * H_DIM * 4, hipMemcpyDeviceToDevice, stream);
  hipMemcpyAsync(Cb, c0, (size_t)B_AGENTS * H_DIM * 4, hipMemcpyDeviceToDevice, stream);

  for (int t = 0; t < PRED_LEN; ++t) {
    k_embed<<<B_AGENTS / RPB, 128, 0, stream>>>(win, ln1w, ln1b, embp, embb, xbuf, t);
    k_cell<<<B_AGENTS / RPB, 256, 0, stream>>>(xbuf, wih_b, whh_b, biasc, Hb, Cb,
                                               ln2w, ln2b, posW, posb,
                                               (float*)d_out, win, t);
  }
}
